// UMLSGraphEmbedding_36206574305712
// MI455X (gfx1250) — matrix operand order repaired and checked by reading, not compiled, not run on hardware
//
#include <hip/hip_runtime.h>
#include <stdint.h>

#define D 128
#define NCUI 100000
#define NVIS 50000
#define ECC 1600000
#define EVC 1000000
#define ECV 1000000
#define NF 2
#define NL 2

typedef __attribute__((ext_vector_type(16))) __bf16 v16bf;
typedef __attribute__((ext_vector_type(8)))  float  v8f;

static __device__ __forceinline__ unsigned short f32_to_bf16_rne(float f) {
    unsigned u = __float_as_uint(f);
    unsigned r = u + 0x7FFFu + ((u >> 16) & 1u);
    return (unsigned short)(r >> 16);
}

// Native CDNA5 fire-and-forget f32 atomic add (no-return -> STOREcnt only,
// drained by S_ENDPGM's implicit wait-idle). Immediate offset folds the
// per-component address math into the VMEM instruction.
static __device__ __forceinline__ void atomic_fadd(float* p, float v) {
    asm volatile("global_atomic_add_f32 %0, %1, off" : : "v"(p), "v"(v) : "memory");
}
static __device__ __forceinline__ void atomic_fadd_off(float* p, float v, int /*imm via template*/) {}

// ---------------- zero (float4) ----------------
__global__ void zero_f32x4(float4* __restrict__ p, long long n4) {
    long long i = (long long)blockIdx.x * blockDim.x + threadIdx.x;
    long long stride = (long long)gridDim.x * blockDim.x;
    float4 z = make_float4(0.f, 0.f, 0.f, 0.f);
    for (; i < n4; i += stride) p[i] = z;
}

// ---------------- scatter-sum: one edge per wave32 ----------------
__global__ void scatter_sum(const float* __restrict__ xsrc,
                            const int* __restrict__ ei, int E,
                            float* __restrict__ agg, float* __restrict__ cnt) {
    int wave = (int)(((long long)blockIdx.x * blockDim.x + threadIdx.x) >> 5);
    int lane = threadIdx.x & 31;
    if (wave >= E) return;
    int s = ei[wave];
    int d = ei[E + wave];
    float4 v = *(const float4*)(xsrc + (long long)s * D + lane * 4);
    float* a = agg + (long long)d * D + lane * 4;
    asm volatile("global_atomic_add_f32 %0, %1, off"           : : "v"(a), "v"(v.x) : "memory");
    asm volatile("global_atomic_add_f32 %0, %1, off offset:4"  : : "v"(a), "v"(v.y) : "memory");
    asm volatile("global_atomic_add_f32 %0, %1, off offset:8"  : : "v"(a), "v"(v.z) : "memory");
    asm volatile("global_atomic_add_f32 %0, %1, off offset:12" : : "v"(a), "v"(v.w) : "memory");
    if (lane == 0) atomic_fadd(cnt + d, 1.0f);
}

// ---------------- pack W' = [Wl | Wr] rows to bf16 hi/lo ----------------
__global__ void pack_weights(const float* __restrict__ Wl, const float* __restrict__ Wr,
                             unsigned short* __restrict__ wh, unsigned short* __restrict__ wlo) {
    int idx = blockIdx.x * blockDim.x + threadIdx.x;
    const int total = NF * NL * 3 * D * 2 * D; // 12 * 128 * 256
    if (idx >= total) return;
    int mat = idx / (D * 2 * D);
    int rem = idx % (D * 2 * D);
    int n   = rem / (2 * D);
    int k   = rem % (2 * D);
    float w = (k < D) ? Wl[((long long)mat * D + n) * D + k]
                      : Wr[((long long)mat * D + n) * D + (k - D)];
    unsigned short h = f32_to_bf16_rne(w);
    float fh = __uint_as_float((unsigned)h << 16);
    wh[idx]  = h;
    wlo[idx] = f32_to_bf16_rne(w - fh);
}

// ---------------- fused SAGE GEMM ----------------
// out[m][n] (init/+=) relu?( (agg[m]/max(cnt,1)) @ Wl^T + b + xdst @ Wr^T )
// Block = 256 thr = 8 waves; 16-row strip; wave w owns 16-col tile n0 = w*16.
// bf16 hi/lo split: A*B ~= Ah*Bh + Ah*Bl + Al*Bh (3x v_wmma_f32_16x16x32_bf16).
__global__ __launch_bounds__(256, 2)
void sage_gemm(const float* __restrict__ agg, const float* __restrict__ cnt,
               const float* __restrict__ xdst,
               const unsigned short* __restrict__ wh, const unsigned short* __restrict__ wlo,
               const float* __restrict__ bias,
               float* __restrict__ out, int nrows, int mode /*0=init,1=add*/, int relu) {
    __shared__ unsigned short lds_hi[16][2 * D];
    __shared__ unsigned short lds_lo[16][2 * D];

    const int m0  = blockIdx.x * 16;
    const int tid = threadIdx.x;

    // ---- stage 16x256 fused A-strip into LDS as bf16 hi/lo ----
    {
        int row = tid >> 4;            // 0..15
        int c0  = (tid & 15) * 16;     // 0..240, each thread does 16 contiguous cols
        int m   = m0 + row;
        float vals[16];
        if (m < nrows) {
            if (c0 < D) {
                float ic = 1.0f / fmaxf(cnt[m], 1.0f);
                const float4* src = (const float4*)(agg + (long long)m * D + c0);
#pragma unroll
                for (int q = 0; q < 4; ++q) {
                    float4 t = src[q];
                    vals[q * 4 + 0] = t.x * ic; vals[q * 4 + 1] = t.y * ic;
                    vals[q * 4 + 2] = t.z * ic; vals[q * 4 + 3] = t.w * ic;
                }
            } else {
                const float4* src = (const float4*)(xdst + (long long)m * D + (c0 - D));
#pragma unroll
                for (int q = 0; q < 4; ++q) {
                    float4 t = src[q];
                    vals[q * 4 + 0] = t.x; vals[q * 4 + 1] = t.y;
                    vals[q * 4 + 2] = t.z; vals[q * 4 + 3] = t.w;
                }
            }
        } else {
#pragma unroll
            for (int q = 0; q < 16; ++q) vals[q] = 0.0f;
        }
#pragma unroll
        for (int q = 0; q < 16; ++q) {
            unsigned short h = f32_to_bf16_rne(vals[q]);
            float fh = __uint_as_float((unsigned)h << 16);
            lds_hi[row][c0 + q] = h;
            lds_lo[row][c0 + q] = f32_to_bf16_rne(vals[q] - fh);
        }
    }
    __syncthreads();

    const int wv   = tid >> 5;
    const int lane = tid & 31;
    const int n0   = wv * 16;
    const int nn   = n0 + (lane & 15);
    const int arow = lane & 15;                 // A: one row per lane (mod 16)
    const int kbA  = (lane < 16) ? 0 : 8;       // A: K interleaved in groups of 8
    const int kbB  = (lane < 16) ? 0 : 8;       // B: K interleaved in groups of 8

    v8f c;
    {
        float bn = bias[nn];
#pragma unroll
        for (int r = 0; r < 8; ++r) c[r] = bn;
    }

    const unsigned short* wrow_h = wh  + (long long)nn * (2 * D);
    const unsigned short* wrow_l = wlo + (long long)nn * (2 * D);

#pragma unroll
    for (int kt = 0; kt < 8; ++kt) {
        int ka = kt * 32 + kbA;
        int kb = kt * 32 + kbB;
        union { v16bf v; uint4 q[2]; } ah, al, bh, bl;
        ah.q[0] = *(const uint4*)&lds_hi[arow][ka];
        ah.q[1] = *(const uint4*)&lds_hi[arow][ka + 16];
        al.q[0] = *(const uint4*)&lds_lo[arow][ka];
        al.q[1] = *(const uint4*)&lds_lo[arow][ka + 16];
        bh.q[0] = *(const uint4*)&wrow_h[kb];
        bh.q[1] = *(const uint4*)&wrow_h[kb + 16];
        bl.q[0] = *(const uint4*)&wrow_l[kb];
        bl.q[1] = *(const uint4*)&wrow_l[kb + 16];
        c = __builtin_amdgcn_wmma_f32_16x16x32_bf16(false, ah.v, false, bh.v, (short)0, c, false, false);
        c = __builtin_amdgcn_wmma_f32_16x16x32_bf16(false, ah.v, false, bl.v, (short)0, c, false, false);
        c = __builtin_amdgcn_wmma_f32_16x16x32_bf16(false, al.v, false, bh.v, (short)0, c, false, false);
    }

    // C/D layout: VGPR r -> M = r (+8 for lanes 16..31), N = lane&15
    const int mbase = m0 + ((lane >> 4) << 3);
#pragma unroll
    for (int r = 0; r < 8; ++r) {
        int m = mbase + r;
        if (m < nrows) {
            long long idx = (long long)m * D + nn;
            float v = c[r];
            if (mode) v += out[idx];
            if (relu) v = fmaxf(v, 0.0f);
            out[idx] = v;
        }
    }
}

// ---------------- elementwise max (filter reduction) ----------------
__global__ void max_inplace(float* __restrict__ out, const float* __restrict__ buf, long long n) {
    long long i = (long long)blockIdx.x * blockDim.x + threadIdx.x;
    long long stride = (long long)gridDim.x * blockDim.x;
    for (; i < n; i += stride) out[i] = fmaxf(out[i], buf[i]);
}

// ---------------- host orchestration ----------------
extern "C" void kernel_launch(void* const* d_in, const int* in_sizes, int n_in,
                              void* d_out, int out_size, void* d_ws, size_t ws_size,
                              hipStream_t stream) {
    const float* x_cui = (const float*)d_in[0];
    const float* x_vis = (const float*)d_in[1];
    const float* W_l   = (const float*)d_in[2];
    const float* b_l   = (const float*)d_in[3];
    const float* W_r   = (const float*)d_in[4];
    const int*   ei_cc = (const int*)d_in[5];
    const int*   ei_vc = (const int*)d_in[6];
    const int*   ei_cv = (const int*)d_in[7];

    // workspace carve-out
    char* ws = (char*)d_ws;
    size_t off = 0;
    auto carve = [&](size_t bytes) -> void* {
        void* p = ws + off;
        off = (off + bytes + 255) & ~(size_t)255;
        return p;
    };
    const size_t WPACK_ELEMS = (size_t)NF * NL * 3 * D * 2 * D; // 393216
    unsigned short* wh   = (unsigned short*)carve(WPACK_ELEMS * 2);
    unsigned short* wlo  = (unsigned short*)carve(WPACK_ELEMS * 2);
    float* agg           = (float*)carve((size_t)NCUI * D * 4);
    float* cnt           = (float*)carve((size_t)NCUI * 4);
    float* bufA_cui      = (float*)carve((size_t)NCUI * D * 4);
    float* bufA_vis      = (float*)carve((size_t)NVIS * D * 4);
    float* bufB_cui      = (float*)carve((size_t)NCUI * D * 4);
    float* bufB_vis      = (float*)carve((size_t)NVIS * D * 4);
    (void)ws_size; (void)in_sizes; (void)n_in; (void)out_size;

    float* out_cui = (float*)d_out;
    float* out_vis = (float*)d_out + (size_t)NCUI * D;

    // pack weights (once per call; deterministic)
    {
        int total = (int)WPACK_ELEMS;
        pack_weights<<<(total + 255) / 256, 256, 0, stream>>>(W_l, W_r, wh, wlo);
    }

    auto run_zero = [&](float* p, long long n) {
        zero_f32x4<<<2048, 256, 0, stream>>>((float4*)p, n >> 2);
    };
    auto run_scatter = [&](const float* src, const int* ei, int E) {
        scatter_sum<<<(E + 7) / 8, 256, 0, stream>>>(src, ei, E, agg, cnt);
    };
    auto run_gemm = [&](const float* xdst, int mat, const float* biasBase,
                        float* out, int nrows, int mode, int relu) {
        const unsigned short* wh_m  = wh  + (size_t)mat * D * 2 * D;
        const unsigned short* wlo_m = wlo + (size_t)mat * D * 2 * D;
        const float* bias = biasBase + (size_t)mat * D;
        sage_gemm<<<(nrows + 15) / 16, 256, 0, stream>>>(
            agg, cnt, xdst, wh_m, wlo_m, bias, out, nrows, mode, relu);
    };

    for (int f = 0; f < NF; ++f) {
        const float* in_cui = x_cui;
        const float* in_vis = x_vis;
        for (int l = 0; l < NL; ++l) {
            float* o_cui;
            float* o_vis;
            if (l < NL - 1) { o_cui = bufA_cui; o_vis = bufA_vis; }
            else if (f == 0) { o_cui = out_cui; o_vis = out_vis; }
            else { o_cui = bufB_cui; o_vis = bufB_vis; }
            int relu = (l < NL - 1) ? 1 : 0;
            int matBase = (f * NL + l) * 3;

            // --- cui output: sage(cc) + sage(vc) ---
            run_zero(agg, (long long)NCUI * D);
            run_zero(cnt, NCUI);
            run_scatter(in_cui, ei_cc, ECC);
            run_gemm(in_cui, matBase + 0, b_l, o_cui, NCUI, /*mode=*/0, /*relu=*/0);

            run_zero(agg, (long long)NCUI * D);
            run_zero(cnt, NCUI);
            run_scatter(in_vis, ei_vc, EVC);
            run_gemm(in_cui, matBase + 1, b_l, o_cui, NCUI, /*mode=*/1, relu);

            // --- vis output: sage(cv) ---
            run_zero(agg, (long long)NVIS * D);
            run_zero(cnt, NVIS);
            run_scatter(in_cui, ei_cv, ECV);
            run_gemm(in_vis, matBase + 2, b_l, o_vis, NVIS, /*mode=*/0, relu);

            in_cui = o_cui;
            in_vis = o_vis;
        }
        if (f == 1) {
            max_inplace<<<2048, 256, 0, stream>>>(out_cui, bufB_cui, (long long)NCUI * D);
            max_inplace<<<2048, 256, 0, stream>>>(out_vis, bufB_vis, (long long)NVIS * D);
        }
    }
}
